// OscillatoryBifurcationGenerator_41540923687082
// MI455X (gfx1250) — matrix operand order repaired and checked by reading, not compiled, not run on hardware
//
#include <hip/hip_runtime.h>
#include <hip/hip_bf16.h>
#include <cstdint>

// Problem constants (reference: B, L, S, H, N, F = 64, 128, 512, 256, 32, 32)
#define BB 64
#define LL 128
#define SS 512
#define HH 256
#define NOSC 32
#define FF 32
#define MM (BB * SS) /* 32768 rows for all big GEMMs */

typedef __attribute__((ext_vector_type(16))) _Float16 v16h;
typedef __attribute__((ext_vector_type(8)))  float    v8f;

__device__ __forceinline__ float lrelu(float x) { return x >= 0.f ? x : 0.2f * x; }

// ---- WMMA operand loaders (wave32, 16x16x32 f16) -------------------------
// A (16x32, MxK): lane l<16 -> row M=l, K {0..7} in v0..3 and {16..23} in v4..7;
// lanes 16..31 -> same rows, K {8..15} and {24..31}.  => per lane: 8 halves at
// (row, ka) and 8 halves at (row, ka+16), ka = (lane>>4)*8.
__device__ __forceinline__ v16h ld_a16(const _Float16* p) {
  union { uint4 u[2]; v16h v; } x;
  x.u[0] = *(const uint4*)(p);
  x.u[1] = *(const uint4*)(p + 16);
  return x.v;
}
// B (32x16, KxN) built from W (N x K row-major, since C = A * W^T):
// lane l -> column N = l&15, 8 K values at kb=(lane>>4)*8 and 8 at kb+16.
__device__ __forceinline__ v16h ld_b16(const _Float16* p) {
  union { uint4 u[2]; v16h v; } x;
  x.u[0] = *(const uint4*)(p);
  x.u[1] = *(const uint4*)(p + 16);
  return x.v;
}
__device__ __forceinline__ v16h zero16() {
  union { uint4 u[2]; v16h v; } x;
  x.u[0] = make_uint4(0u, 0u, 0u, 0u);
  x.u[1] = make_uint4(0u, 0u, 0u, 0u);
  return x.v;
}
__device__ __forceinline__ v8f wmma_f16(v16h a, v16h b, v8f c) {
  return __builtin_amdgcn_wmma_f32_16x16x32_f16(false, a, false, b, (short)0, c,
                                                false, false);
}

// ---- Generic WMMA GEMM: out = epilogue(A(MxK,f16) * W(NxK,f16)^T + bias) ---
// K is compile-time -> fully unrolled, software-pipelined (double-buffered
// A + B registers) so loads for step k+1 issue before the wmmas of step k.
enum { EP_F32 = 0, EP_LRELU_F16 = 1, EP_RESID = 2, EP_TANH_OUT = 3 };

template <int NTILES, int MODE, int K>
__global__ __launch_bounds__(256) void gemm_wmma(
    const _Float16* __restrict__ A, const _Float16* __restrict__ W,
    const float* __restrict__ bias, float* __restrict__ out32,
    _Float16* __restrict__ out16, const float* __restrict__ res32,
    int M_, int N_) {
  const int lane = threadIdx.x & 31;
  const int wave = (blockIdx.x * blockDim.x + threadIdx.x) >> 5;
  const int ngroups = N_ / (16 * NTILES);
  const int mt = wave / ngroups;
  const int ng = wave - mt * ngroups;
  if (mt * 16 >= M_) return;
  const int l15 = lane & 15;
  const int ka = (lane >> 4) * 8;
  const int kb = (lane >> 4) * 8;
  const _Float16* Ap = A + (size_t)(mt * 16 + l15) * K + ka;
  const _Float16* Wp = W + (size_t)(ng * (16 * NTILES) + l15) * K + kb;

  constexpr int STEPS = K / 32;
  v8f acc[NTILES] = {};

  v16h a_cur = ld_a16(Ap);
  v16h b_cur[NTILES];
#pragma unroll
  for (int t = 0; t < NTILES; ++t) b_cur[t] = ld_b16(Wp + (size_t)t * 16 * K);

#pragma unroll
  for (int s = 0; s < STEPS; ++s) {
    v16h a_nxt;
    v16h b_nxt[NTILES];
    if (s + 1 < STEPS) {
      a_nxt = ld_a16(Ap + (s + 1) * 32);
#pragma unroll
      for (int t = 0; t < NTILES; ++t)
        b_nxt[t] = ld_b16(Wp + (size_t)t * 16 * K + (s + 1) * 32);
    }
#pragma unroll
    for (int t = 0; t < NTILES; ++t) acc[t] = wmma_f16(a_cur, b_cur[t], acc[t]);
    if (s + 1 < STEPS) {
      a_cur = a_nxt;
#pragma unroll
      for (int t = 0; t < NTILES; ++t) b_cur[t] = b_nxt[t];
    }
  }

  // C/D layout: lane holds col N = lane&15; VGPR v holds row M = v + (lane>>4)*8
  const int rb = mt * 16 + ((lane >> 4) << 3);
#pragma unroll
  for (int t = 0; t < NTILES; ++t) {
    const int col = ng * (16 * NTILES) + t * 16 + l15;
    const float bv = bias[col];
#pragma unroll
    for (int v = 0; v < 8; ++v) {
      const size_t idx = (size_t)(rb + v) * N_ + col;
      float x = acc[t][v] + bv;
      if (MODE == EP_LRELU_F16) {
        x = lrelu(x);
        out16[idx] = (_Float16)x;
      } else if (MODE == EP_RESID) {
        const float h = res32[idx] + 0.1f * x;
        out32[idx] = h;
        out16[idx] = (_Float16)h;
      } else if (MODE == EP_TANH_OUT) {
        out32[idx] = tanhf(x);
      } else {
        out32[idx] = x;
      }
    }
  }
}

// ---- Conv1d(k=3, pad 1) over sequence as 3 shifted WMMA GEMMs -------------
// Per k-step: batch all 12 B loads, prefetch next step's 3 A tiles, then 12
// wmmas (graduated s_wait_loadcnt gives intra-step load/compute overlap).
__global__ __launch_bounds__(256) void conv_gemm_wmma(
    const _Float16* __restrict__ A, const _Float16* __restrict__ W0,
    const _Float16* __restrict__ W1, const _Float16* __restrict__ W2,
    const float* __restrict__ bias, float* __restrict__ out32) {
  const int lane = threadIdx.x & 31;
  const int wave = (blockIdx.x * blockDim.x + threadIdx.x) >> 5;
  const int mt = wave >> 2;  // 4 n-groups of 64 cols
  const int ng = wave & 3;
  const int l15 = lane & 15;
  const int ka = (lane >> 4) * 8;
  const int kb = (lane >> 4) * 8;
  const int arow = mt * 16 + l15;
  const int s = arow & (SS - 1);
  const bool vm = (s >= 1), vp = (s <= SS - 2);
  const _Float16* Ap = A + (size_t)arow * HH + ka;
  const size_t wbase = (size_t)(ng * 64 + l15) * HH + kb;

  v8f acc[4] = {};
  v16h am = vm ? ld_a16(Ap - HH) : zero16();
  v16h a0 = ld_a16(Ap);
  v16h ap = vp ? ld_a16(Ap + HH) : zero16();

#pragma unroll
  for (int st = 0; st < HH / 32; ++st) {
    const int k0 = st * 32;
    // all B tiles for this k-step
    v16h b[12];
#pragma unroll
    for (int t = 0; t < 4; ++t) {
      const size_t wo = wbase + (size_t)t * 16 * HH + k0;
      b[t * 3 + 0] = ld_b16(W0 + wo);
      b[t * 3 + 1] = ld_b16(W1 + wo);
      b[t * 3 + 2] = ld_b16(W2 + wo);
    }
    // prefetch next step's A tiles
    v16h amn, a0n, apn;
    if (st + 1 < HH / 32) {
      const int k1 = k0 + 32;
      amn = vm ? ld_a16(Ap + k1 - HH) : zero16();
      a0n = ld_a16(Ap + k1);
      apn = vp ? ld_a16(Ap + k1 + HH) : zero16();
    }
#pragma unroll
    for (int t = 0; t < 4; ++t) {
      acc[t] = wmma_f16(am, b[t * 3 + 0], acc[t]);
      acc[t] = wmma_f16(a0, b[t * 3 + 1], acc[t]);
      acc[t] = wmma_f16(ap, b[t * 3 + 2], acc[t]);
    }
    if (st + 1 < HH / 32) { am = amn; a0 = a0n; ap = apn; }
  }

  const int rb = mt * 16 + ((lane >> 4) << 3);
#pragma unroll
  for (int t = 0; t < 4; ++t) {
    const int col = ng * 64 + t * 16 + l15;
    const float bv = bias[col];
#pragma unroll
    for (int v = 0; v < 8; ++v)
      out32[(size_t)(rb + v) * HH + col] = acc[t][v] + bv;
  }
}

// ---- Per-(batch,channel) norm over S + lrelu + residual 0.1 ---------------
__global__ __launch_bounds__(256) void conv_norm_resid(
    const float* __restrict__ c, float* __restrict__ hf32,
    _Float16* __restrict__ hf16) {
  const int b = blockIdx.x, o = threadIdx.x;  // 256 channels
  float s = 0.f, s2 = 0.f;
  for (int si = 0; si < SS; ++si) {
    const float v = c[((size_t)b * SS + si) * HH + o];
    s += v; s2 += v * v;
  }
  const float m = s * (1.f / SS);
  const float var = s2 * (1.f / SS) - m * m;
  const float rinv = rsqrtf(var + 1e-5f);
  for (int si = 0; si < SS; ++si) {
    const size_t idx = ((size_t)b * SS + si) * HH + o;
    const float v = lrelu((c[idx] - m) * rinv);
    const float h = hf32[idx] + 0.1f * v;
    hf32[idx] = h;
    hf16[idx] = (_Float16)h;
  }
}

// ---- Fused 3-layer input MLP (linear+LN+lrelu x3), one block per batch ----
__device__ __forceinline__ void bred256(float* r1, float* r2, float a, float b,
                                        float& s, float& s2) {
  const int t = threadIdx.x;
  __syncthreads();
  r1[t] = a; r2[t] = b;
  __syncthreads();
  for (int ofs = 128; ofs > 0; ofs >>= 1) {
    if (t < ofs) { r1[t] += r1[t + ofs]; r2[t] += r2[t + ofs]; }
    __syncthreads();
  }
  s = r1[0]; s2 = r2[0];
}

__global__ __launch_bounds__(256) void mlp3_kernel(
    const float* __restrict__ z,
    const float* __restrict__ w1, const float* __restrict__ b1,
    const float* __restrict__ g1, const float* __restrict__ be1,
    const float* __restrict__ w2, const float* __restrict__ b2,
    const float* __restrict__ g2, const float* __restrict__ be2,
    const float* __restrict__ w3, const float* __restrict__ b3,
    const float* __restrict__ g3, const float* __restrict__ be3,
    float* __restrict__ hbase) {
  __shared__ float zx[LL];
  __shared__ float x1[4 * HH];
  __shared__ float x2[2 * HH];
  __shared__ float r1[256], r2[256];
  const int b = blockIdx.x, t = threadIdx.x;
  if (t < LL) zx[t] = z[b * LL + t];
  __syncthreads();

  // layer 1: 1024 out, 4 per thread
  float o[4];
#pragma unroll
  for (int j = 0; j < 4; ++j) {
    const int oo = t * 4 + j;
    const float* wr = w1 + (size_t)oo * LL;
    float s = b1[oo];
    for (int i = 0; i < LL; ++i) s += zx[i] * wr[i];
    o[j] = s;
  }
  float su = 0.f, sq = 0.f, m, v2;
#pragma unroll
  for (int j = 0; j < 4; ++j) { su += o[j]; sq += o[j] * o[j]; }
  bred256(r1, r2, su, sq, su, sq);
  m = su * (1.f / 1024.f);
  float rinv = rsqrtf(sq * (1.f / 1024.f) - m * m + 1e-5f);
#pragma unroll
  for (int j = 0; j < 4; ++j) {
    const int oo = t * 4 + j;
    x1[oo] = lrelu((o[j] - m) * rinv * g1[oo] + be1[oo]);
  }
  __syncthreads();

  // layer 2: 512 out, 2 per thread
  const float* wr0 = w2 + (size_t)t * 1024;
  const float* wr1 = w2 + (size_t)(t + 256) * 1024;
  float q0 = b2[t], q1 = b2[t + 256];
  for (int i = 0; i < 1024; ++i) {
    const float xv = x1[i];
    q0 += xv * wr0[i]; q1 += xv * wr1[i];
  }
  bred256(r1, r2, q0 + q1, q0 * q0 + q1 * q1, su, sq);
  m = su * (1.f / 512.f);
  rinv = rsqrtf(sq * (1.f / 512.f) - m * m + 1e-5f);
  x2[t] = lrelu((q0 - m) * rinv * g2[t] + be2[t]);
  x2[t + 256] = lrelu((q1 - m) * rinv * g2[t + 256] + be2[t + 256]);
  __syncthreads();

  // layer 3: 256 out, 1 per thread
  const float* wr = w3 + (size_t)t * 512;
  float s3 = b3[t];
  for (int i = 0; i < 512; ++i) s3 += x2[i] * wr[i];
  bred256(r1, r2, s3, s3 * s3, su, sq);
  m = su * (1.f / 256.f);
  rinv = rsqrtf(sq * (1.f / 256.f) - m * m + 1e-5f);
  v2 = (s3 - m) * rinv * g3[t] + be3[t];
  hbase[b * HH + t] = lrelu(v2);
}

// ---- Broadcast over S + sinusoidal positional encoding --------------------
__global__ __launch_bounds__(256) void bcast_pe(const float* __restrict__ hbase,
                                                float* __restrict__ hf32,
                                                _Float16* __restrict__ hf16) {
  const size_t idx = (size_t)blockIdx.x * blockDim.x + threadIdx.x;  // < M*H
  const int e = (int)(idx & (HH - 1));
  const int bs = (int)(idx >> 8);
  const int s = bs & (SS - 1);
  const int b = bs >> 9;
  const int i2 = e >> 1;
  const float div = __expf((float)(2 * i2) * (-9.210340371976184f / (float)HH));
  const float ang = (float)s * div;
  const float pe = (e & 1) ? __cosf(ang) : __sinf(ang);
  const float v = hbase[b * HH + e] + pe;
  hf32[idx] = v;
  hf16[idx] = (_Float16)v;
}

// ---- Oscillator bifurcation step: one wave per (b,s), lane = oscillator ---
__global__ __launch_bounds__(256) void osc_step(
    const float* __restrict__ osc, const float* __restrict__ freqs,
    const float* __restrict__ coupling, const float* __restrict__ tm_w,
    const float* __restrict__ tm_b, const float* __restrict__ mu_p,
    _Float16* __restrict__ comb) {
  const int lane = threadIdx.x & 31;
  const int wid = (blockIdx.x * blockDim.x + threadIdx.x) >> 5;  // = b*S + s
  const int s = wid & (SS - 1);
  const int j = lane;
  if (s == 0) {  // keep first timestep unchanged
    comb[(size_t)wid * 64 + j] = (_Float16)osc[(size_t)wid * 64 + j];
    comb[(size_t)wid * 64 + 32 + j] = (_Float16)osc[(size_t)wid * 64 + 32 + j];
    return;
  }
  const float* prev = osc + (size_t)(wid - 1) * 64;
  const float p = prev[j];
  const float a = prev[32 + j];
  float rowsum = 0.f;
  for (int k = 0; k < NOSC; ++k) rowsum += coupling[j * NOSC + k];
  float csum = 0.f;
  for (int k = 0; k < NOSC; ++k) {
    const float pk = __shfl(p, k, 32);
    const float rk = __shfl(rowsum, k, 32);
    csum += __sinf(p - pk) * rk;
  }
  const float tmod = (float)s * tm_w[j] + tm_b[j];
  const float dph = freqs[j] + csum * 0.1f + tmod * 0.1f;  // COUP = 0.1
  const float pn = p + dph * 0.1f;
  const float mu = mu_p[0];
  const float damp = (mu - 0.1f) * a - a * a * a;  // DECAY = 0.1
  const float an = tanhf(a + damp * 0.1f);          // SAT = 1.0
  comb[(size_t)wid * 64 + j] = (_Float16)pn;
  comb[(size_t)wid * 64 + 32 + j] = (_Float16)an;
}

// ---- LayerNorm(128) + lrelu -> f16, one wave per row ----------------------
__global__ __launch_bounds__(256) void ln128_lrelu(
    const float* __restrict__ x, const float* __restrict__ g,
    const float* __restrict__ be, _Float16* __restrict__ out) {
  const int lane = threadIdx.x & 31;
  const int row = (blockIdx.x * blockDim.x + threadIdx.x) >> 5;
  const float* xr = x + (size_t)row * 128;
  float v[4];
  float s = 0.f, s2 = 0.f;
#pragma unroll
  for (int q = 0; q < 4; ++q) {
    v[q] = xr[lane + q * 32];
    s += v[q]; s2 += v[q] * v[q];
  }
  for (int o2 = 16; o2 > 0; o2 >>= 1) {
    s += __shfl_xor(s, o2, 32);
    s2 += __shfl_xor(s2, o2, 32);
  }
  const float m = s * (1.f / 128.f);
  const float rinv = rsqrtf(s2 * (1.f / 128.f) - m * m + 1e-5f);
  _Float16* orow = out + (size_t)row * 128;
#pragma unroll
  for (int q = 0; q < 4; ++q) {
    const int f = lane + q * 32;
    orow[f] = (_Float16)lrelu((v[q] - m) * rinv * g[f] + be[f]);
  }
}

// ---- Weight precision prep ------------------------------------------------
__global__ void cvt16(const float* __restrict__ src, _Float16* __restrict__ dst,
                      int n) {
  const int i = blockIdx.x * blockDim.x + threadIdx.x;
  if (i < n) dst[i] = (_Float16)src[i];
}
// conv_w is (O, I, 3) -> three (O, I) row-major f16 matrices
__global__ void conv_repack(const float* __restrict__ w, _Float16* __restrict__ w0,
                            _Float16* __restrict__ w1, _Float16* __restrict__ w2) {
  const int i = blockIdx.x * blockDim.x + threadIdx.x;  // 65536
  const int o = i >> 8, ii = i & 255;
  const float* p = w + (size_t)o * (HH * 3) + ii * 3;
  w0[i] = (_Float16)p[0];
  w1[i] = (_Float16)p[1];
  w2[i] = (_Float16)p[2];
}

extern "C" void kernel_launch(void* const* d_in, const int* in_sizes, int n_in,
                              void* d_out, int out_size, void* d_ws,
                              size_t ws_size, hipStream_t stream) {
  (void)in_sizes; (void)n_in; (void)out_size; (void)ws_size;
#define IN(i) ((const float*)d_in[(i)])
  const float* z = IN(0);
  const float *np_w1 = IN(1), *np_b1 = IN(2), *np_g1 = IN(3), *np_be1 = IN(4);
  const float *np_w2 = IN(5), *np_b2 = IN(6), *np_g2 = IN(7), *np_be2 = IN(8);
  const float *np_w3 = IN(9), *np_b3 = IN(10), *np_g3 = IN(11), *np_be3 = IN(12);
  const float* conv_w[2] = {IN(13), IN(15)};
  const float* conv_b[2] = {IN(14), IN(16)};
  const float *mu[2], *freqs[2], *coupling[2], *to_w1[2], *to_b1[2], *to_w2[2],
      *to_b2[2], *fr_w1[2], *fr_b1[2], *fr_w2[2], *fr_b2[2], *tm_w[2], *tm_b[2];
  for (int i = 0; i < 2; ++i) {
    const int bs = 17 + 13 * i;
    mu[i] = IN(bs + 0); freqs[i] = IN(bs + 1); coupling[i] = IN(bs + 2);
    to_w1[i] = IN(bs + 3); to_b1[i] = IN(bs + 4);
    to_w2[i] = IN(bs + 5); to_b2[i] = IN(bs + 6);
    fr_w1[i] = IN(bs + 7); fr_b1[i] = IN(bs + 8);
    fr_w2[i] = IN(bs + 9); fr_b2[i] = IN(bs + 10);
    tm_w[i] = IN(bs + 11); tm_b[i] = IN(bs + 12);
  }
  const float *op_w1 = IN(43), *op_b1 = IN(44), *op_g1 = IN(45),
              *op_be1 = IN(46), *op_w2 = IN(47), *op_b2 = IN(48);

  // --- workspace carve (~115 MB) ---
  char* ws = (char*)d_ws;
  size_t off = 0;
  auto carve = [&](size_t bytes) -> void* {
    void* p = ws + off;
    off += (bytes + 255) & ~(size_t)255;
    return p;
  };
  float* hf32 = (float*)carve((size_t)MM * HH * 4);
  float* tmp32 = (float*)carve((size_t)MM * HH * 4);
  _Float16* hf16 = (_Float16*)carve((size_t)MM * HH * 2);
  _Float16* tmp16 = (_Float16*)carve((size_t)MM * HH * 2);
  float* osc32 = (float*)carve((size_t)MM * 64 * 4);
  _Float16* comb16 = (_Float16*)carve((size_t)MM * 64 * 2);
  float* hbase = (float*)carve((size_t)BB * HH * 4);
  _Float16* cw[2][3];
  for (int l = 0; l < 2; ++l)
    for (int k = 0; k < 3; ++k) cw[l][k] = (_Float16*)carve(65536 * 2);
  _Float16 *tw1_16[2], *tw2_16[2], *fw1_16[2], *fw2_16[2];
  for (int i = 0; i < 2; ++i) {
    tw1_16[i] = (_Float16*)carve(65536 * 2);
    tw2_16[i] = (_Float16*)carve(16384 * 2);
    fw1_16[i] = (_Float16*)carve(16384 * 2);
    fw2_16[i] = (_Float16*)carve(65536 * 2);
  }
  _Float16* op1_16 = (_Float16*)carve(32768 * 2);
  _Float16* op2_16 = (_Float16*)carve(4096 * 2);

  // --- weight prep (f32 -> f16) ---
  for (int l = 0; l < 2; ++l)
    conv_repack<<<65536 / 256, 256, 0, stream>>>(conv_w[l], cw[l][0], cw[l][1], cw[l][2]);
  for (int i = 0; i < 2; ++i) {
    cvt16<<<65536 / 256, 256, 0, stream>>>(to_w1[i], tw1_16[i], 65536);
    cvt16<<<16384 / 256, 256, 0, stream>>>(to_w2[i], tw2_16[i], 16384);
    cvt16<<<16384 / 256, 256, 0, stream>>>(fr_w1[i], fw1_16[i], 16384);
    cvt16<<<65536 / 256, 256, 0, stream>>>(fr_w2[i], fw2_16[i], 65536);
  }
  cvt16<<<32768 / 256, 256, 0, stream>>>(op_w1, op1_16, 32768);
  cvt16<<<4096 / 256, 256, 0, stream>>>(op_w2, op2_16, 4096);

  // --- input MLP + broadcast/pos-encoding ---
  mlp3_kernel<<<BB, 256, 0, stream>>>(z, np_w1, np_b1, np_g1, np_be1, np_w2,
                                      np_b2, np_g2, np_be2, np_w3, np_b3,
                                      np_g3, np_be3, hbase);
  bcast_pe<<<(MM * HH) / 256, 256, 0, stream>>>(hbase, hf32, hf16);

  // --- 2 conv blocks: 3-shift WMMA GEMM then norm-over-S + residual ---
  for (int l = 0; l < 2; ++l) {
    conv_gemm_wmma<<<1024, 256, 0, stream>>>(hf16, cw[l][0], cw[l][1], cw[l][2],
                                             conv_b[l], tmp32);
    conv_norm_resid<<<BB, 256, 0, stream>>>(tmp32, hf32, hf16);
  }

  // --- 2 bifurcation layers ---
  for (int i = 0; i < 2; ++i) {
    gemm_wmma<4, EP_LRELU_F16, HH><<<1024, 256, 0, stream>>>(
        hf16, tw1_16[i], to_b1[i], nullptr, tmp16, nullptr, MM, HH);
    gemm_wmma<4, EP_F32, HH><<<256, 256, 0, stream>>>(
        tmp16, tw2_16[i], to_b2[i], osc32, nullptr, nullptr, MM, 64);
    osc_step<<<MM / 8, 256, 0, stream>>>(osc32, freqs[i], coupling[i], tm_w[i],
                                         tm_b[i], mu[i], comb16);
    gemm_wmma<4, EP_LRELU_F16, 64><<<1024, 256, 0, stream>>>(
        comb16, fw1_16[i], fr_b1[i], nullptr, tmp16, nullptr, MM, HH);
    gemm_wmma<4, EP_RESID, HH><<<1024, 256, 0, stream>>>(
        tmp16, fw2_16[i], fr_b2[i], hf32, hf16, hf32, MM, HH);
  }

  // --- output head ---
  gemm_wmma<4, EP_F32, HH><<<512, 256, 0, stream>>>(
      hf16, op1_16, op_b1, tmp32, nullptr, nullptr, MM, 128);
  ln128_lrelu<<<MM / 8, 256, 0, stream>>>(tmp32, op_g1, op_be1, tmp16);
  gemm_wmma<2, EP_TANH_OUT, 128><<<256, 256, 0, stream>>>(
      tmp16, op2_16, op_b2, (float*)d_out, nullptr, nullptr, MM, 32);
#undef IN
}
